// DecoderBlock_12678743458364
// MI455X (gfx1250) — matrix operand order repaired and checked by reading, not compiled, not run on hardware
//
#include <hip/hip_runtime.h>
#include <stdint.h>

// ---------------------------------------------------------------------------
// Feature detection: async global->LDS copies (CDNA5, ASYNCcnt-tracked)
// ---------------------------------------------------------------------------
#if defined(__has_builtin)
#  if __has_builtin(__builtin_amdgcn_global_load_async_to_lds_b128)
#    define USE_ASYNC_LDS 1
#  endif
#endif
#ifndef USE_ASYNC_LDS
#  define USE_ASYNC_LDS 0
#endif

typedef __attribute__((ext_vector_type(4))) int v4i;

#if USE_ASYNC_LDS
typedef __attribute__((address_space(1))) v4i GASInt4;   // global int4
typedef __attribute__((address_space(3))) v4i LASInt4;   // LDS int4

__device__ __forceinline__ void g2l_async_b128(const void* g, void* l) {
  // per-lane 16B: global -> LDS copy  (GLOBAL_LOAD_ASYNC_TO_LDS_B128)
  __builtin_amdgcn_global_load_async_to_lds_b128(
      (GASInt4*)(uintptr_t)g, (LASInt4*)(uint32_t)(uintptr_t)l, 0, 0);
}
__device__ __forceinline__ void wait_async0() {
#  if __has_builtin(__builtin_amdgcn_s_wait_asynccnt)
  __builtin_amdgcn_s_wait_asynccnt(0);
#  else
  asm volatile("s_wait_asynccnt 0x0" ::: "memory");
#  endif
}
#endif

// ---------------------------------------------------------------------------
// Types / helpers
// ---------------------------------------------------------------------------
typedef __attribute__((ext_vector_type(16))) __bf16 bf16x16;
typedef __attribute__((ext_vector_type(8)))  float  f32x8;

union FragB16 {
  bf16x16  v;
  uint16_t u[16];
  uint4    q[2];
};

__device__ __forceinline__ uint16_t f2bf(float f) {
  uint32_t x = __float_as_uint(f);
  x += 0x7FFFu + ((x >> 16) & 1u);   // round-to-nearest-even
  return (uint16_t)(x >> 16);
}

__device__ __forceinline__ f32x8 wmma_bf16(bf16x16 a, bf16x16 b, f32x8 c) {
  // D = A(16x32) * B(32x16) + C, fp32 accumulate
  return __builtin_amdgcn_wmma_f32_16x16x32_bf16(false, a, false, b,
                                                 (short)0, c, false, false);
}

// A-fragment (16x32 bf16) from a row-major [rows][32] bf16 tile.
// Lane l: row m = mbase + (l&15), half h = l>>4.
// Elements 0..7 cover K = 8h..8h+7 ; elements 8..15 cover K = 16+8h..16+8h+7.
__device__ __forceinline__ bf16x16 load_a_frag32(const uint16_t* t, int mbase, int lane) {
  int m = mbase + (lane & 15);
  int h = lane >> 4;
  const uint16_t* p = t + m * 32;
  FragB16 f;
  f.q[0] = *(const uint4*)(p + 8 * h);
  f.q[1] = *(const uint4*)(p + 16 + 8 * h);
  return f.v;
}

// B-fragment (32x16 bf16) from a row-major [32][ldn] bf16 tile.
// Lane l holds column N = nbase+(l&15); elements 0..7 K = 8h.., 8..15 K = 16+8h.. (h = l>>4).
__device__ __forceinline__ bf16x16 load_b_frag(const uint16_t* t, int nbase, int ldn, int lane) {
  const uint16_t* p = t + nbase + (lane & 15);
  const int hh = lane >> 4;
  FragB16 f;
#pragma unroll
  for (int e = 0; e < 8; ++e) {
    f.u[e]     = p[(8 * hh + e) * ldn];
    f.u[8 + e] = p[(16 + 8 * hh + e) * ldn];
  }
  return f.v;
}

// ---------------------------------------------------------------------------
// fp32 -> bf16 conversion
// ---------------------------------------------------------------------------
__global__ __launch_bounds__(256) void cvt_f32_bf16(const float* __restrict__ in,
                                                    uint16_t* __restrict__ out, int n) {
  int i = blockIdx.x * 256 + threadIdx.x;
  if (i < n) out[i] = f2bf(in[i]);
}

// ---------------------------------------------------------------------------
// LayerNorm: fp32 [rows][Dn] -> bf16, one 256-thread block per row (Dn = 1024)
// ---------------------------------------------------------------------------
__global__ __launch_bounds__(256) void layernorm_bf16(const float* __restrict__ x,
                                                      const float* __restrict__ g,
                                                      const float* __restrict__ b,
                                                      uint16_t* __restrict__ out, int Dn) {
  int row = blockIdx.x, tid = threadIdx.x;
  const float* xr = x + (size_t)row * Dn;
  float4 v = *(const float4*)(xr + tid * 4);
  float s  = v.x + v.y + v.z + v.w;
  float s2 = v.x * v.x + v.y * v.y + v.z * v.z + v.w * v.w;
#pragma unroll
  for (int m = 1; m < 32; m <<= 1) {
    s  += __shfl_xor(s,  m, 32);
    s2 += __shfl_xor(s2, m, 32);
  }
  __shared__ float red[2][8];
  int wave = tid >> 5, lane = tid & 31;
  if (lane == 0) { red[0][wave] = s; red[1][wave] = s2; }
  __syncthreads();
  float ts = 0.f, ts2 = 0.f;
#pragma unroll
  for (int i = 0; i < 8; ++i) { ts += red[0][i]; ts2 += red[1][i]; }
  float mu   = ts / (float)Dn;
  float var  = ts2 / (float)Dn - mu * mu;
  float rstd = rsqrtf(var + 1e-5f);
  int c = tid * 4;
  uint16_t* o = out + (size_t)row * Dn + c;
  o[0] = f2bf((v.x - mu) * rstd * g[c + 0] + b[c + 0]);
  o[1] = f2bf((v.y - mu) * rstd * g[c + 1] + b[c + 1]);
  o[2] = f2bf((v.z - mu) * rstd * g[c + 2] + b[c + 2]);
  o[3] = f2bf((v.w - mu) * rstd * g[c + 3] + b[c + 3]);
}

// ---------------------------------------------------------------------------
// Generic WMMA GEMM: C[M,N] = A_bf16[M,K] @ W_bf16[K,N] + bias (+GELU) (+resid)
// Block tile 128x128, BK=32, 8 waves (2 in M x 4 in N), 64x32 per wave.
// Double-buffered LDS; tiles fed by GLOBAL_LOAD_ASYNC_TO_LDS_B128 when the
// toolchain exposes it (ASYNCcnt-synchronized), else register-staged loads.
// ---------------------------------------------------------------------------
__global__ __launch_bounds__(256) void gemm_bf16_wmma(
    const uint16_t* __restrict__ A, const uint16_t* __restrict__ W,
    const float* __restrict__ bias, const float* __restrict__ resid,
    float* __restrict__ Cf, uint16_t* __restrict__ Cb,
    int M, int N, int K, int gelu) {
  __shared__ alignas(16) uint16_t sA[2][128 * 32];
  __shared__ alignas(16) uint16_t sB[2][32 * 128];

  const int tid  = threadIdx.x;
  const int lane = tid & 31;
  const int wave = tid >> 5;
  const int bm   = blockIdx.y * 128;
  const int bn   = blockIdx.x * 128;
  const int wm   = (wave & 1) * 64;   // wave row offset in block tile
  const int wn   = (wave >> 1) * 32;  // wave col offset in block tile

  // per-thread tile-copy slice
  const int ar = tid >> 1, ak = (tid & 1) * 16;       // A: 128 rows x 32 k
  const int br = tid >> 3, bnn = (tid & 7) * 16;      // B: 32 k-rows x 128 n

#if USE_ASYNC_LDS
  auto copy_tile = [&](int kt, int buf) {
    const uint16_t* ap = A + (size_t)(bm + ar) * K + kt * 32 + ak;
    g2l_async_b128(ap,     &sA[buf][ar * 32 + ak]);
    g2l_async_b128(ap + 8, &sA[buf][ar * 32 + ak + 8]);
    const uint16_t* bp = W + (size_t)(kt * 32 + br) * N + bn + bnn;
    g2l_async_b128(bp,     &sB[buf][br * 128 + bnn]);
    g2l_async_b128(bp + 8, &sB[buf][br * 128 + bnn + 8]);
  };
#else
  uint4 ra0, ra1, rb0, rb1;
  auto gload = [&](int kt) {
    const uint16_t* ap = A + (size_t)(bm + ar) * K + kt * 32 + ak;
    ra0 = *(const uint4*)(ap);
    ra1 = *(const uint4*)(ap + 8);
    const uint16_t* bp = W + (size_t)(kt * 32 + br) * N + bn + bnn;
    rb0 = *(const uint4*)(bp);
    rb1 = *(const uint4*)(bp + 8);
  };
  auto lstore = [&](int buf) {
    *(uint4*)(&sA[buf][ar * 32 + ak])       = ra0;
    *(uint4*)(&sA[buf][ar * 32 + ak + 8])   = ra1;
    *(uint4*)(&sB[buf][br * 128 + bnn])     = rb0;
    *(uint4*)(&sB[buf][br * 128 + bnn + 8]) = rb1;
  };
#endif

  f32x8 acc[4][2] = {};

#if USE_ASYNC_LDS
  copy_tile(0, 0);
  wait_async0();
#else
  gload(0);
  lstore(0);
#endif
  __syncthreads();

  const int nk = K / 32;
  for (int kt = 0; kt < nk; ++kt) {
    const int cur = kt & 1;
#if USE_ASYNC_LDS
    if (kt + 1 < nk) copy_tile(kt + 1, cur ^ 1);  // async copy runs under WMMA
#else
    if (kt + 1 < nk) gload(kt + 1);               // overlap next-tile fetch
#endif

    bf16x16 af[4], bfm[2];
#pragma unroll
    for (int i = 0; i < 4; ++i) af[i] = load_a_frag32(sA[cur], wm + 16 * i, lane);
#pragma unroll
    for (int j = 0; j < 2; ++j) bfm[j] = load_b_frag(sB[cur], wn + 16 * j, 128, lane);
#pragma unroll
    for (int i = 0; i < 4; ++i)
#pragma unroll
      for (int j = 0; j < 2; ++j)
        acc[i][j] = wmma_bf16(af[i], bfm[j], acc[i][j]);

#if USE_ASYNC_LDS
    if (kt + 1 < nk) wait_async0();
#else
    if (kt + 1 < nk) lstore(cur ^ 1);
#endif
    __syncthreads();
  }

  // Epilogue: bias (+GELU) (+residual), store fp32 and/or bf16.
  const int lrow = 8 * (lane >> 4);
  const int lcol = lane & 15;
#pragma unroll
  for (int i = 0; i < 4; ++i)
#pragma unroll
    for (int j = 0; j < 2; ++j)
#pragma unroll
      for (int v = 0; v < 8; ++v) {
        int gm = bm + wm + 16 * i + v + lrow;
        int gn = bn + wn + 16 * j + lcol;
        float val = acc[i][j][v] + bias[gn];
        if (gelu) val = 0.5f * val * (1.0f + erff(val * 0.70710678118654752f));
        if (resid) val += resid[(size_t)gm * N + gn];
        if (Cf) Cf[(size_t)gm * N + gn] = val;
        if (Cb) Cb[(size_t)gm * N + gn] = f2bf(val);
      }
}

// ---------------------------------------------------------------------------
// Flash attention (causal), bf16 Q/K/V laid out as [B*T, D] with head columns.
// Block = 256 threads = 8 waves; each wave handles a 16-query tile of one head.
// Per 32-key block: 4 WMMA for S = Q K^T, 4 WMMA for O += P V.
// ---------------------------------------------------------------------------
__global__ __launch_bounds__(256) void flash_attn_wmma(
    const uint16_t* __restrict__ Q, const uint16_t* __restrict__ Km,
    const uint16_t* __restrict__ V, uint16_t* __restrict__ O,
    int Tn, int Dn, float scale) {
  __shared__ alignas(16) uint16_t lds_p[8 * 16 * 32];  // per-wave 16x32 P tile

  const int tid  = threadIdx.x;
  const int lane = tid & 31;
  const int wave = tid >> 5;
  const int h    = blockIdx.y;
  const int bidx = blockIdx.z;
  const int qs   = (blockIdx.x * 8 + wave) * 16;

  const size_t bT   = (size_t)bidx * Tn;
  const int    hoff = h * 64;
  const int    m16  = lane & 15;
  const int    half = lane >> 4;

  // Q fragments (two d-steps of 32)
  FragB16 qf[2];
  {
    const uint16_t* qrow = Q + (bT + qs + m16) * Dn + hoff;
#pragma unroll
    for (int s = 0; s < 2; ++s) {
      qf[s].q[0] = *(const uint4*)(qrow + s * 32 + 8 * half);
      qf[s].q[1] = *(const uint4*)(qrow + s * 32 + 16 + 8 * half);
    }
  }

  f32x8 o[4] = {};
  float mi[8], li[8];
#pragma unroll
  for (int v = 0; v < 8; ++v) { mi[v] = -INFINITY; li[v] = 0.f; }

  const int nkb = (qs + 16 + 31) >> 5;  // causal: only key blocks <= query tile
  for (int kb = 0; kb < nkb; ++kb) {
    const int kbase = kb * 32;

    // K^T B-fragments: lane -> key j*16+m16 ; elements -> d in table order (s*32 + 8*half.., 16+8*half..)
    FragB16 ktf[2][2];
#pragma unroll
    for (int j = 0; j < 2; ++j)
#pragma unroll
      for (int s = 0; s < 2; ++s) {
        const uint16_t* kp = Km + (bT + kbase + j * 16 + m16) * Dn + hoff + s * 32;
        ktf[j][s].q[0] = *(const uint4*)(kp + 8 * half);
        ktf[j][s].q[1] = *(const uint4*)(kp + 16 + 8 * half);
      }

    f32x8 S[2] = {};
#pragma unroll
    for (int j = 0; j < 2; ++j)
#pragma unroll
      for (int s = 0; s < 2; ++s)
        S[j] = wmma_bf16(qf[s].v, ktf[j][s].v, S[j]);

    // scale + causal mask
#pragma unroll
    for (int j = 0; j < 2; ++j)
#pragma unroll
      for (int v = 0; v < 8; ++v) {
        int key  = kbase + j * 16 + m16;
        int qrow = qs + v + 8 * half;
        float x  = S[j][v] * scale;
        S[j][v]  = (key > qrow) ? -INFINITY : x;
      }

    // online softmax (row reductions across the 16-lane half-groups)
    float alpha[8];
#pragma unroll
    for (int v = 0; v < 8; ++v) {
      float t = fmaxf(S[0][v], S[1][v]);
#pragma unroll
      for (int m = 1; m < 16; m <<= 1) t = fmaxf(t, __shfl_xor(t, m, 32));
      float mnew = fmaxf(mi[v], t);
      alpha[v]   = __expf(mi[v] - mnew);  // exp(-inf)=0 on first block
      mi[v]      = mnew;
    }
#pragma unroll
    for (int j = 0; j < 2; ++j)
#pragma unroll
      for (int v = 0; v < 8; ++v) S[j][v] = __expf(S[j][v] - mi[v]);
#pragma unroll
    for (int v = 0; v < 8; ++v) {
      float t = S[0][v] + S[1][v];
#pragma unroll
      for (int m = 1; m < 16; m <<= 1) t += __shfl_xor(t, m, 32);
      li[v] = li[v] * alpha[v] + t;
    }
#pragma unroll
    for (int f = 0; f < 4; ++f)
#pragma unroll
      for (int v = 0; v < 8; ++v) o[f][v] *= alpha[v];

    // P (C-layout) -> LDS -> A-fragment layout (same-wave, DScnt-ordered)
    uint16_t* lp = lds_p + wave * (16 * 32);
#pragma unroll
    for (int j = 0; j < 2; ++j)
#pragma unroll
      for (int v = 0; v < 8; ++v)
        lp[(v + 8 * half) * 32 + j * 16 + m16] = f2bf(S[j][v]);
    bf16x16 pf = load_a_frag32(lp, 0, lane);

    // V B-fragments: lane -> d column f*16+m16; elements -> keys in table order
#pragma unroll
    for (int f = 0; f < 4; ++f) {
      const uint16_t* vrow = V + (bT + kbase) * Dn + hoff + f * 16 + m16;
      FragB16 vf;
#pragma unroll
      for (int e = 0; e < 8; ++e) {
        vf.u[e]     = vrow[(size_t)(8 * half + e) * Dn];
        vf.u[8 + e] = vrow[(size_t)(16 + 8 * half + e) * Dn];
      }
      o[f] = wmma_bf16(pf, vf.v, o[f]);
    }
  }

  // epilogue: O /= l, store bf16
#pragma unroll
  for (int v = 0; v < 8; ++v) {
    float inv = 1.0f / li[v];
    int row = qs + v + 8 * half;
#pragma unroll
    for (int f = 0; f < 4; ++f)
      O[(bT + row) * Dn + hoff + f * 16 + m16] = f2bf(o[f][v] * inv);
  }
}

// ---------------------------------------------------------------------------
// Host orchestration
// ---------------------------------------------------------------------------
extern "C" void kernel_launch(void* const* d_in, const int* in_sizes, int n_in,
                              void* d_out, int out_size, void* d_ws, size_t ws_size,
                              hipStream_t stream) {
  constexpr int Bn = 4, Tn = 1024, Dn = 1024, Hn = 16, DFF = 4096;
  constexpr int Mn = Bn * Tn;  // 4096

  const float* x_enc = (const float*)d_in[0];
  const float* x_in  = (const float*)d_in[1];
  const float* ln1_g = (const float*)d_in[2];
  const float* ln1_b = (const float*)d_in[3];
  const float* ln2_g = (const float*)d_in[4];
  const float* ln2_b = (const float*)d_in[5];
  const float* sa_wq = (const float*)d_in[6];  const float* sa_bq = (const float*)d_in[7];
  const float* sa_wk = (const float*)d_in[8];  const float* sa_bk = (const float*)d_in[9];
  const float* sa_wv = (const float*)d_in[10]; const float* sa_bv = (const float*)d_in[11];
  const float* sa_wo = (const float*)d_in[12]; const float* sa_bo = (const float*)d_in[13];
  const float* ca_wq = (const float*)d_in[14]; const float* ca_bq = (const float*)d_in[15];
  const float* ca_wk = (const float*)d_in[16]; const float* ca_bk = (const float*)d_in[17];
  const float* ca_wv = (const float*)d_in[18]; const float* ca_bv = (const float*)d_in[19];
  const float* ca_wo = (const float*)d_in[20]; const float* ca_bo = (const float*)d_in[21];
  const float* mlp_w1 = (const float*)d_in[22]; const float* mlp_b1 = (const float*)d_in[23];
  const float* mlp_w2 = (const float*)d_in[24]; const float* mlp_b2 = (const float*)d_in[25];

  uint8_t* ws = (uint8_t*)d_ws;
  size_t off = 0;
  auto take = [&](size_t bytes) -> uint8_t* {
    uint8_t* p = ws + off;
    off = (off + bytes + 255) & ~(size_t)255;
    return p;
  };
  const size_t DD = (size_t)Dn * Dn, MD = (size_t)Mn * Dn;
  uint16_t* wsaq = (uint16_t*)take(DD * 2);
  uint16_t* wsak = (uint16_t*)take(DD * 2);
  uint16_t* wsav = (uint16_t*)take(DD * 2);
  uint16_t* wsao = (uint16_t*)take(DD * 2);
  uint16_t* wcaq = (uint16_t*)take(DD * 2);
  uint16_t* wcak = (uint16_t*)take(DD * 2);
  uint16_t* wcav = (uint16_t*)take(DD * 2);
  uint16_t* wcao = (uint16_t*)take(DD * 2);
  uint16_t* w1b  = (uint16_t*)take((size_t)Dn * DFF * 2);
  uint16_t* w2b  = (uint16_t*)take((size_t)DFF * Dn * 2);
  uint16_t* xencb = (uint16_t*)take(MD * 2);
  uint16_t* lnb   = (uint16_t*)take(MD * 2);
  uint16_t* Qb    = (uint16_t*)take(MD * 2);
  uint16_t* Kb    = (uint16_t*)take(MD * 2);
  uint16_t* Vb    = (uint16_t*)take(MD * 2);
  uint16_t* attb  = (uint16_t*)take(MD * 2);
  float*    x1    = (float*)take(MD * 4);
  float*    x2    = (float*)take(MD * 4);
  uint16_t* h1b   = (uint16_t*)take((size_t)Mn * DFF * 2);

  auto cvt = [&](const float* src, uint16_t* dst, size_t n) {
    cvt_f32_bf16<<<dim3((unsigned)((n + 255) / 256)), dim3(256), 0, stream>>>(src, dst, (int)n);
  };
  auto gemm = [&](const uint16_t* A, const uint16_t* W, const float* bias,
                  const float* resid, float* Cf, uint16_t* Cb, int M, int N, int K, int gelu) {
    gemm_bf16_wmma<<<dim3(N / 128, M / 128), dim3(256), 0, stream>>>(
        A, W, bias, resid, Cf, Cb, M, N, K, gelu);
  };
  auto ln = [&](const float* src, const float* g, const float* b, uint16_t* dst) {
    layernorm_bf16<<<dim3(Mn), dim3(256), 0, stream>>>(src, g, b, dst, Dn);
  };
  auto attn = [&](const uint16_t* Qp, const uint16_t* Kp, const uint16_t* Vp, uint16_t* Op) {
    flash_attn_wmma<<<dim3(Tn / 128, Hn, Bn), dim3(256), 0, stream>>>(
        Qp, Kp, Vp, Op, Tn, Dn, 0.125f);
  };

  // weight / encoder conversions to bf16
  cvt(sa_wq, wsaq, DD); cvt(sa_wk, wsak, DD); cvt(sa_wv, wsav, DD); cvt(sa_wo, wsao, DD);
  cvt(ca_wq, wcaq, DD); cvt(ca_wk, wcak, DD); cvt(ca_wv, wcav, DD); cvt(ca_wo, wcao, DD);
  cvt(mlp_w1, w1b, (size_t)Dn * DFF); cvt(mlp_w2, w2b, (size_t)DFF * Dn);
  cvt(x_enc, xencb, MD);

  // ---- self attention: x1 = x + SA(ln1(x)) ----
  ln(x_in, ln1_g, ln1_b, lnb);
  gemm(lnb, wsaq, sa_bq, nullptr, nullptr, Qb, Mn, Dn, Dn, 0);
  gemm(lnb, wsak, sa_bk, nullptr, nullptr, Kb, Mn, Dn, Dn, 0);
  gemm(lnb, wsav, sa_bv, nullptr, nullptr, Vb, Mn, Dn, Dn, 0);
  attn(Qb, Kb, Vb, attb);
  gemm(attb, wsao, sa_bo, x_in, x1, nullptr, Mn, Dn, Dn, 0);

  // ---- cross attention (causal, ln1 reused): x2 = x1 + CA(ln1(x1), x_enc) ----
  ln(x1, ln1_g, ln1_b, lnb);
  gemm(lnb,   wcaq, ca_bq, nullptr, nullptr, Qb, Mn, Dn, Dn, 0);
  gemm(xencb, wcak, ca_bk, nullptr, nullptr, Kb, Mn, Dn, Dn, 0);
  gemm(xencb, wcav, ca_bv, nullptr, nullptr, Vb, Mn, Dn, Dn, 0);
  attn(Qb, Kb, Vb, attb);
  gemm(attb, wcao, ca_bo, x1, x2, nullptr, Mn, Dn, Dn, 0);

  // ---- MLP: out = x2 + (gelu(ln2(x2) @ w1 + b1) @ w2 + b2) ----
  ln(x2, ln2_g, ln2_b, lnb);
  gemm(lnb, w1b, mlp_b1, nullptr, nullptr, h1b, Mn, DFF, Dn, 1);
  gemm(h1b, w2b, mlp_b2, x2, (float*)d_out, nullptr, Mn, Dn, DFF, 0);
}
